// Yolact_72756745994449
// MI455X (gfx1250) — matrix operand order repaired and checked by reading, not compiled, not run on hardware
//
#include <hip/hip_runtime.h>
#include <hip/hip_bf16.h>

// ---------------------------------------------------------------------------
// YOLACT head for MI455X (gfx1250, wave32, WMMA).
// Convs are implicit GEMMs on V_WMMA_F32_16X16X32_F16:
//   M = Cout (tile 16), N = Hout*Wout, K = Cin*kh*kw (chunks of 32).
// Block = 4 waves; weight (A) fragment staged once per block in LDS in
// fragment-lane order, shared by all waves; each wave drives 2 accumulators
// (16x32 strip) so one A fragment feeds 2 WMMA issues per K-chunk.
// Kernel window size is a template parameter (1 or 9) so the K->(ci,r,s)
// decomposition is compile-time (no runtime division in the hot loop).
// ---------------------------------------------------------------------------

typedef __attribute__((ext_vector_type(16))) _Float16 v16h;
typedef __attribute__((ext_vector_type(8)))  float    v8f;

#define NUM_CLASSES 81
#define MASK_DIM 32
#define NUM_PRIORS 3
#define TOTAL_ANCHORS 19248

#define CONV_WAVES 4
#define CONV_NPW   2                       // 16x16 tiles per wave
#define CONV_BLKPIX (CONV_WAVES * CONV_NPW * 16)   // 128 pixels per block

// ---------------------------------------------------------------------------
// Generic conv (NCHW in, OIHW weights) as implicit GEMM on WMMA.
// blockIdx.x = 128-pixel tile, blockIdx.y = 16-out-channel tile, z = batch.
// Requires: Ktot = Cin*KHKW is a multiple of 32 (true for all convs here:
// 256, 512, 1024, 2304). For KHKW==1 the conv must be stride-1 / pad-0.
// ---------------------------------------------------------------------------
template <int KHKW>
__global__ __launch_bounds__(128)
void conv_wmma_kernel(const float* __restrict__ in,
                      const float* __restrict__ wgt,
                      const float* __restrict__ bias,
                      float* __restrict__ out,
                      int Cin, int H, int W,
                      int Cout, int Hout, int Wout,
                      int stride, int pad, int relu)
{
    constexpr int KW = (KHKW == 9) ? 3 : 1;

    // A tile staged in fragment-lane order: row = lane, cols 0..15 = v16h halves.
    // 64-byte row stride keeps each lane's 32-byte fragment load aligned.
    __shared__ alignas(64) _Float16 aLds[32][32];

    const int tid  = threadIdx.x;
    const int lane = tid & 31;
    const int wave = tid >> 5;
    const int n    = lane & 15;            // B-fragment col (N)
    const int hi   = (lane >= 16) ? 1 : 0;

    const int HWo  = Hout * Wout;
    const int oc0  = blockIdx.y * 16;
    const int Ktot = Cin * KHKW;

    const float* inN = in + (size_t)blockIdx.z * (size_t)Cin * H * W;

    v8f acc[CONV_NPW];
    #pragma unroll
    for (int j = 0; j < CONV_NPW; ++j) acc[j] = (v8f){0.f,0.f,0.f,0.f,0.f,0.f,0.f,0.f};

    // per-sub-tile pixel coordinates (top-left of receptive field, pre-padded)
    int pg[CONV_NPW], ohs[CONV_NPW], ows[CONV_NPW];
    bool valid[CONV_NPW];
    #pragma unroll
    for (int j = 0; j < CONV_NPW; ++j) {
        pg[j] = blockIdx.x * CONV_BLKPIX + wave * (CONV_NPW * 16) + j * 16 + n;
        valid[j] = pg[j] < HWo;
        int oh = 0, ow = 0;
        if (valid[j]) { oh = pg[j] / Wout; ow = pg[j] % Wout; }
        ohs[j] = oh * stride - pad;
        ows[j] = ow * stride - pad;
    }

    for (int k0 = 0; k0 < Ktot; k0 += 32) {
        // ---- cooperative fill of A tile (16x32 weights -> fragment order) ----
        #pragma unroll
        for (int e = 0; e < 4; ++e) {
            int idx = tid * 4 + e;                 // 0..511
            int l   = idx >> 4;                    // destination lane
            int t   = idx & 15;                    // half index within v16h
            int lm  = l & 15;
            int lhi = (l >= 16) ? 1 : 0;
            // ISA 7.12.2 16-bit A layout: lanes 0-15 halves -> K {0..7,16..23};
            // lanes 16-31 -> K {8..15,24..31}
            int ka = k0 + (lhi ? ((t < 8) ? t + 8 : t + 16)
                               : ((t < 8) ? t     : t + 8));
            int mrow = oc0 + lm;
            float av = 0.f;
            if (mrow < Cout) {
                av = wgt[(size_t)mrow * Ktot + ka];
                if (ka + 32 < Ktot)                 // prefetch next K-chunk
                    __builtin_prefetch(&wgt[(size_t)mrow * Ktot + ka + 32], 0, 1);
            }
            aLds[l][t] = (_Float16)av;
        }
        __syncthreads();

        // ---- each lane pulls its full A fragment with one 32B LDS read ----
        v16h a = *(const v16h*)(&aLds[lane][0]);

        // ---- B fragments: activations, 32x16 f16 layout; 2 tiles reuse A ----
        #pragma unroll
        for (int j = 0; j < CONV_NPW; ++j) {
            v16h b;
            #pragma unroll
            for (int t = 0; t < 16; ++t) {
                int kb = k0 + (hi ? 8 : 0) + ((t < 8) ? t : t + 8);
                float bv = 0.f;
                if (valid[j]) {
                    if (KHKW == 1) {
                        // 1x1, stride 1, pad 0: always in range
                        bv = inN[((size_t)kb * H + ohs[j]) * W + ows[j]];
                    } else {
                        int ci  = kb / KHKW;            // compile-time divisor
                        int rem = kb - ci * KHKW;
                        int r   = rem / KW;
                        int s   = rem - r * KW;
                        int ih  = ohs[j] + r;
                        int iw  = ows[j] + s;
                        if (ih >= 0 && ih < H && iw >= 0 && iw < W)
                            bv = inN[((size_t)ci * H + ih) * W + iw];
                    }
                }
                b[t] = (_Float16)bv;
            }
            acc[j] = __builtin_amdgcn_wmma_f32_16x16x32_f16(
                         false, a, false, b, (short)0, acc[j], false, false);
        }
        __syncthreads();
    }

    // ---- C/D store: VGPR r, lane l -> (M = r + 8*hi, N = l&15) ----
    float* outN = out + (size_t)blockIdx.z * (size_t)Cout * HWo;
    #pragma unroll
    for (int j = 0; j < CONV_NPW; ++j) {
        #pragma unroll
        for (int r = 0; r < 8; ++r) {
            int mrow = oc0 + r + (hi ? 8 : 0);
            if (mrow < Cout && valid[j]) {
                float v = acc[j][r] + bias[mrow];
                if (relu) v = fmaxf(v, 0.f);
                outN[(size_t)mrow * HWo + pg[j]] = v;
            }
        }
    }
}

// ---------------------------------------------------------------------------
// Half-pixel bilinear resize over `planes` independent 2D planes.
// addMode=1: dst += resize(src); addMode=0: dst = (relu? max(0,.) : .)
// ---------------------------------------------------------------------------
__global__ void bilinear_kernel(const float* __restrict__ src,
                                float* __restrict__ dst,
                                int planes, int Hs, int Ws, int Hd, int Wd,
                                int addMode, int relu)
{
    long long idx = (long long)blockIdx.x * blockDim.x + threadIdx.x;
    long long total = (long long)planes * Hd * Wd;
    if (idx >= total) return;
    int wd = (int)(idx % Wd);
    int hd = (int)((idx / Wd) % Hd);
    int pl = (int)(idx / ((long long)Wd * Hd));

    float sy = (hd + 0.5f) * (float)Hs / (float)Hd - 0.5f;
    float sx = (wd + 0.5f) * (float)Ws / (float)Wd - 0.5f;
    sy = fminf(fmaxf(sy, 0.f), (float)(Hs - 1));
    sx = fminf(fmaxf(sx, 0.f), (float)(Ws - 1));
    int y0 = (int)sy, x0 = (int)sx;
    int y1 = min(y0 + 1, Hs - 1);
    int x1 = min(x0 + 1, Ws - 1);
    float fy = sy - (float)y0, fx = sx - (float)x0;

    const float* p = src + (size_t)pl * Hs * Ws;
    float v00 = p[(size_t)y0 * Ws + x0];
    float v01 = p[(size_t)y0 * Ws + x1];
    float v10 = p[(size_t)y1 * Ws + x0];
    float v11 = p[(size_t)y1 * Ws + x1];
    float v = v00 * (1.f - fy) * (1.f - fx) + v01 * (1.f - fy) * fx
            + v10 * fy * (1.f - fx)         + v11 * fy * fx;

    if (addMode) dst[idx] += v;
    else         dst[idx] = relu ? fmaxf(v, 0.f) : v;
}

// ---------------------------------------------------------------------------
// Head post-processing (full batch): NCHW conv outputs -> [B, anchors, dim]
// with the reference's transpose(0,2,3,1).reshape ordering:
// anchor = lvlOff + pix*3 + prior.
// ---------------------------------------------------------------------------
__global__ void loc_post_kernel(const float* __restrict__ src, float* __restrict__ dst,
                                int HW, int lvlOff, int B)
{
    int idx = blockIdx.x * blockDim.x + threadIdx.x;     // over B*HW*12
    if (idx >= B * HW * 12) return;
    int b   = idx / (HW * 12);
    int rem = idx % (HW * 12);
    int pix = rem / 12;
    int r2  = rem % 12;
    int pr = r2 / 4, j = r2 % 4;
    float v = src[(size_t)b * 12 * HW + (size_t)(pr * 4 + j) * HW + pix];
    dst[((size_t)b * TOTAL_ANCHORS + lvlOff + (size_t)pix * 3 + pr) * 4 + j] = v;
}

__global__ void conf_post_kernel(const float* __restrict__ src, float* __restrict__ dst,
                                 int HW, int lvlOff, int B)
{
    int idx = blockIdx.x * blockDim.x + threadIdx.x;     // over B*HW*3 anchors
    if (idx >= B * HW * 3) return;
    int b   = idx / (HW * 3);
    int rem = idx % (HW * 3);
    int pix = rem / 3, pr = rem % 3;
    const float* s0 = src + (size_t)b * NUM_PRIORS * NUM_CLASSES * HW;
    float mx = -3.4e38f;
    for (int c = 0; c < NUM_CLASSES; ++c)
        mx = fmaxf(mx, s0[(size_t)(pr * NUM_CLASSES + c) * HW + pix]);
    float s = 0.f;
    for (int c = 0; c < NUM_CLASSES; ++c)
        s += expf(s0[(size_t)(pr * NUM_CLASSES + c) * HW + pix] - mx);
    float inv = 1.f / s;
    float* o = dst + ((size_t)b * TOTAL_ANCHORS + lvlOff + (size_t)pix * 3 + pr) * NUM_CLASSES;
    for (int c = 0; c < NUM_CLASSES; ++c)
        o[c] = expf(s0[(size_t)(pr * NUM_CLASSES + c) * HW + pix] - mx) * inv;
}

__global__ void mask_post_kernel(const float* __restrict__ src, float* __restrict__ dst,
                                 int HW, int lvlOff, int B)
{
    int idx = blockIdx.x * blockDim.x + threadIdx.x;     // over B*HW*96
    if (idx >= B * HW * NUM_PRIORS * MASK_DIM) return;
    int b   = idx / (HW * NUM_PRIORS * MASK_DIM);
    int rem = idx % (HW * NUM_PRIORS * MASK_DIM);
    int pix = rem / (NUM_PRIORS * MASK_DIM);
    int r2  = rem % (NUM_PRIORS * MASK_DIM);
    int pr = r2 / MASK_DIM, c = r2 % MASK_DIM;
    float v = tanhf(src[(size_t)b * NUM_PRIORS * MASK_DIM * HW
                        + (size_t)(pr * MASK_DIM + c) * HW + pix]);
    dst[((size_t)b * TOTAL_ANCHORS + lvlOff + (size_t)pix * 3 + pr) * MASK_DIM + c] = v;
}

// proto: NCHW (32,138,138) single image -> NHWC with ReLU
__global__ void proto_post_kernel(const float* __restrict__ src, float* __restrict__ dst,
                                  int HW, int b)
{
    int idx = blockIdx.x * blockDim.x + threadIdx.x;     // over HW*32
    if (idx >= HW * MASK_DIM) return;
    int pix = idx / MASK_DIM, c = idx % MASK_DIM;
    float v = fmaxf(src[(size_t)c * HW + pix], 0.f);
    dst[((size_t)b * HW + pix) * MASK_DIM + c] = v;
}

__global__ void priors_kernel(float* __restrict__ dst, int H, int W,
                              float scale, int lvlOff)
{
    int idx = blockIdx.x * blockDim.x + threadIdx.x;     // over H*W*3
    if (idx >= H * W * 3) return;
    int pix = idx / 3, ai = idx % 3;
    int y = pix / W, x = pix % W;
    float cx = (x + 0.5f) / (float)W;
    float cy = (y + 0.5f) / (float)H;
    float ar = (ai == 0) ? 1.0f : ((ai == 1) ? 0.5f : 2.0f);
    float wh = scale * sqrtf(ar) / 550.0f;
    float* o = dst + (size_t)(lvlOff + idx) * 4;
    o[0] = cx; o[1] = cy; o[2] = wh; o[3] = wh;
}

// ---------------------------------------------------------------------------
// Host orchestration
// ---------------------------------------------------------------------------
extern "C" void kernel_launch(void* const* d_in, const int* in_sizes, int n_in,
                              void* d_out, int out_size, void* d_ws, size_t ws_size,
                              hipStream_t stream)
{
    (void)in_sizes; (void)n_in; (void)out_size; (void)ws_size;
    const float* c3 = (const float*)d_in[0];   // [8,256,69,69]
    const float* c4 = (const float*)d_in[1];   // [8,512,35,35]
    const float* c5 = (const float*)d_in[2];   // [8,1024,18,18]
    // params in dict insertion order
    const float* lat0_w = (const float*)d_in[3];  const float* lat0_b = (const float*)d_in[4];
    const float* lat1_w = (const float*)d_in[5];  const float* lat1_b = (const float*)d_in[6];
    const float* lat2_w = (const float*)d_in[7];  const float* lat2_b = (const float*)d_in[8];
    const float* fp0_w  = (const float*)d_in[9];  const float* fp0_b  = (const float*)d_in[10];
    const float* fp1_w  = (const float*)d_in[11]; const float* fp1_b  = (const float*)d_in[12];
    const float* fp2_w  = (const float*)d_in[13]; const float* fp2_b  = (const float*)d_in[14];
    const float* dn0_w  = (const float*)d_in[15]; const float* dn0_b  = (const float*)d_in[16];
    const float* dn1_w  = (const float*)d_in[17]; const float* dn1_b  = (const float*)d_in[18];
    const float* pr0_w  = (const float*)d_in[19]; const float* pr0_b  = (const float*)d_in[20];
    const float* pr1_w  = (const float*)d_in[21]; const float* pr1_b  = (const float*)d_in[22];
    const float* pr2_w  = (const float*)d_in[23]; const float* pr2_b  = (const float*)d_in[24];
    const float* pr3_w  = (const float*)d_in[25]; const float* pr3_b  = (const float*)d_in[26];
    const float* pr4_w  = (const float*)d_in[27]; const float* pr4_b  = (const float*)d_in[28];
    const float* up_w   = (const float*)d_in[29]; const float* up_b   = (const float*)d_in[30];
    const float* bb_w   = (const float*)d_in[31]; const float* bb_b   = (const float*)d_in[32];
    const float* cf_w   = (const float*)d_in[33]; const float* cf_b   = (const float*)d_in[34];
    const float* mk_w   = (const float*)d_in[35]; const float* mk_b   = (const float*)d_in[36];

    float* ws  = (float*)d_ws;
    float* out = (float*)d_out;

    const int B = 8;
    // workspace layout (floats)
    const size_t O_LAT5 = 0;                              // 8*256*324
    const size_t O_LAT4 = O_LAT5 + 663552;                // 8*256*1225
    const size_t O_LAT3 = O_LAT4 + 2508800;               // 8*256*4761
    const size_t O_P5   = O_LAT3 + 9748992;
    const size_t O_P4   = O_P5   + 663552;
    const size_t O_P3   = O_P4   + 2508800;
    const size_t O_P6   = O_P3   + 9748992;               // 8*256*81
    const size_t O_P7   = O_P6   + 165888;                // 8*256*25
    const size_t O_T1   = O_P7   + 51200;                 // 256*4761 (per-image)
    const size_t O_T2   = O_T1   + 1218816;
    const size_t O_U1   = O_T2   + 1218816;               // 256*19044 (per-image)
    const size_t O_U2   = O_U1   + 4875264;
    const size_t O_HU   = O_U2   + 4875264;               // 8*256*4761 (full batch)
    const size_t O_HB   = O_HU   + 9748992;               // 8*12*4761
    const size_t O_HC   = O_HB   + 457056;                // 8*243*4761
    const size_t O_HM   = O_HC   + 9255384;               // 8*96*4761

    // d_out layout (floats)
    const size_t D_LOC    = 0;                            // 8*19248*4
    const size_t D_CONF   = D_LOC  + (size_t)B * TOTAL_ANCHORS * 4;
    const size_t D_MASK   = D_CONF + (size_t)B * TOTAL_ANCHORS * NUM_CLASSES;
    const size_t D_PRIORS = D_MASK + (size_t)B * TOTAL_ANCHORS * MASK_DIM;
    const size_t D_PROTO  = D_PRIORS + (size_t)TOTAL_ANCHORS * 4;

    auto conv = [&](const float* in_, const float* w_, const float* b_, float* out_,
                    int N, int Cin, int H, int W, int Cout,
                    int k, int stride, int pad, int relu) {
        int Hout = (H + 2 * pad - k) / stride + 1;
        int Wout = (W + 2 * pad - k) / stride + 1;
        dim3 g((unsigned)((Hout * Wout + CONV_BLKPIX - 1) / CONV_BLKPIX),
               (unsigned)((Cout + 15) / 16), (unsigned)N);
        if (k == 1)
            conv_wmma_kernel<1><<<g, 128, 0, stream>>>(in_, w_, b_, out_,
                Cin, H, W, Cout, Hout, Wout, stride, pad, relu);
        else
            conv_wmma_kernel<9><<<g, 128, 0, stream>>>(in_, w_, b_, out_,
                Cin, H, W, Cout, Hout, Wout, stride, pad, relu);
    };
    auto resize = [&](const float* src, float* dst, int planes,
                      int Hs, int Ws, int Hd, int Wd, int addMode, int relu) {
        long long total = (long long)planes * Hd * Wd;
        int blocks = (int)((total + 255) / 256);
        bilinear_kernel<<<blocks, 256, 0, stream>>>(src, dst, planes, Hs, Ws, Hd, Wd, addMode, relu);
    };

    // ---------------- FPN (full batch) ----------------
    conv(c5, lat0_w, lat0_b, ws + O_LAT5, B, 1024, 18, 18, 256, 1, 1, 0, 0);
    conv(c4, lat1_w, lat1_b, ws + O_LAT4, B,  512, 35, 35, 256, 1, 1, 0, 0);
    conv(c3, lat2_w, lat2_b, ws + O_LAT3, B,  256, 69, 69, 256, 1, 1, 0, 0);
    resize(ws + O_LAT5, ws + O_LAT4, B * 256, 18, 18, 35, 35, /*add*/1, 0);  // x = resize(p5)+lat(c4)
    resize(ws + O_LAT4, ws + O_LAT3, B * 256, 35, 35, 69, 69, /*add*/1, 0);  // x = resize(x)+lat(c3)
    conv(ws + O_LAT5, fp0_w, fp0_b, ws + O_P5, B, 256, 18, 18, 256, 3, 1, 1, 1);
    conv(ws + O_LAT4, fp1_w, fp1_b, ws + O_P4, B, 256, 35, 35, 256, 3, 1, 1, 1);
    conv(ws + O_LAT3, fp2_w, fp2_b, ws + O_P3, B, 256, 69, 69, 256, 3, 1, 1, 1);
    conv(ws + O_P5, dn0_w, dn0_b, ws + O_P6, B, 256, 18, 18, 256, 3, 2, 1, 0);  // -> 9x9
    conv(ws + O_P6, dn1_w, dn1_b, ws + O_P7, B, 256,  9,  9, 256, 3, 2, 1, 0);  // -> 5x5

    // ---------------- proto net on P3 (per image; 138^2 buffers are large) ---
    for (int b = 0; b < B; ++b) {
        const float* p3b = ws + O_P3 + (size_t)b * 256 * 4761;
        conv(p3b,        pr0_w, pr0_b, ws + O_T1, 1, 256, 69, 69, 256, 3, 1, 1, 1);
        conv(ws + O_T1,  pr1_w, pr1_b, ws + O_T2, 1, 256, 69, 69, 256, 3, 1, 1, 1);
        conv(ws + O_T2,  pr2_w, pr2_b, ws + O_T1, 1, 256, 69, 69, 256, 3, 1, 1, 1);
        resize(ws + O_T1, ws + O_U1, 256, 69, 69, 138, 138, /*add*/0, /*relu*/1);
        conv(ws + O_U1,  pr3_w, pr3_b, ws + O_U2, 1, 256, 138, 138, 256, 3, 1, 1, 1);
        conv(ws + O_U2,  pr4_w, pr4_b, ws + O_U1, 1, 256, 138, 138,  32, 1, 1, 0, 0);
        int hw = 138 * 138;
        proto_post_kernel<<<(hw * MASK_DIM + 255) / 256, 256, 0, stream>>>(
            ws + O_U1, out + D_PROTO, hw, b);
    }

    // ---------------- shared prediction heads (full batch, per level) --------
    const int   lvlH[5]   = {69, 35, 18, 9, 5};
    const int   lvlOff[5] = {0, 14283, 17958, 18930, 19173};
    const float lvlScale[5] = {24.f, 48.f, 96.f, 192.f, 384.f};
    const size_t featOff[5] = {O_P3, O_P4, O_P5, O_P6, O_P7};

    for (int l = 0; l < 5; ++l) {
        int H = lvlH[l], HW = H * H;
        const float* f = ws + featOff[l];
        conv(f,          up_w, up_b, ws + O_HU, B, 256, H, H, 256,                      3, 1, 1, 1);
        conv(ws + O_HU,  bb_w, bb_b, ws + O_HB, B, 256, H, H, NUM_PRIORS * 4,           3, 1, 1, 0);
        conv(ws + O_HU,  cf_w, cf_b, ws + O_HC, B, 256, H, H, NUM_PRIORS * NUM_CLASSES, 3, 1, 1, 0);
        conv(ws + O_HU,  mk_w, mk_b, ws + O_HM, B, 256, H, H, NUM_PRIORS * MASK_DIM,    3, 1, 1, 0);
        loc_post_kernel<<<(B * HW * 12 + 255) / 256, 256, 0, stream>>>(
            ws + O_HB, out + D_LOC, HW, lvlOff[l], B);
        conf_post_kernel<<<(B * HW * 3 + 255) / 256, 256, 0, stream>>>(
            ws + O_HC, out + D_CONF, HW, lvlOff[l], B);
        mask_post_kernel<<<(B * HW * NUM_PRIORS * MASK_DIM + 255) / 256, 256, 0, stream>>>(
            ws + O_HM, out + D_MASK, HW, lvlOff[l], B);
    }

    // ---------------- priors ----------------
    for (int l = 0; l < 5; ++l) {
        int H = lvlH[l];
        priors_kernel<<<(H * H * 3 + 255) / 256, 256, 0, stream>>>(
            out + D_PRIORS, H, H, lvlScale[l], lvlOff[l]);
    }
    (void)O_HM;
}
